// self_attention_1374389534912
// MI455X (gfx1250) — matrix operand order repaired and checked by reading, not compiled, not run on hardware
//
#include <hip/hip_runtime.h>

// Problem dims
constexpr int S_DIM = 2048;
constexpr int D_DIM = 512;
constexpr int B_DIM = 8;

// GEMM tiling: 128x128 block tile, 8 waves (wave32) arranged 4(M) x 2(N);
// each wave owns a 32x64 tile = 2x4 WMMA fragments -> 8 v_wmma per K-step.
// LDS double-buffered: fragment loads (cur) -> global fetch (next) -> WMMA ->
// LDS commit (next). One barrier per K-step.
constexpr int BM = 128;
constexpr int BN = 128;
constexpr int BK = 32;
constexpr int LDA = BK + 4;  // padded LDS row (36 halves = 72B): 8B-aligned, 18-dword stride
constexpr int LDB = BK + 4;

typedef __attribute__((ext_vector_type(16))) _Float16 v16h;
typedef __attribute__((ext_vector_type(8)))  float    v8f;

union FragH { v16h v; unsigned u[8]; };
union H4 { uint2 u; _Float16 h[4]; };

__device__ __forceinline__ v8f wmma_f16(v16h a, v16h b, v8f c) {
  return __builtin_amdgcn_wmma_f32_16x16x32_f16(false, a, false, b, (short)0, c,
                                                false, false);
}

// A fragment (16x32 f16, M x K). Lanes 0-15: M=lane; lanes 16-31: same M, K+8.
__device__ __forceinline__ v16h load_frag_a(const _Float16* As, int mbase, int lane) {
  FragH r;
  const unsigned* p =
      reinterpret_cast<const unsigned*>(As + (mbase + (lane & 15)) * LDA);
  const int half = lane >> 4;
#pragma unroll
  for (int v = 0; v < 8; ++v) {
    const int k = ((v >> 2) << 4) | (half << 3) | ((v & 3) << 1);
    r.u[v] = p[k >> 1];
  }
  return r.v;
}

// B fragment (32x16 f16, K x N), LDS stored N-major (Bst[n][k]).
__device__ __forceinline__ v16h load_frag_b(const _Float16* Bst, int nbase, int lane) {
  FragH r;
  const unsigned* p =
      reinterpret_cast<const unsigned*>(Bst + (nbase + (lane & 15)) * LDB);
  const int khalf = lane >> 4;
#pragma unroll
  for (int v = 0; v < 8; ++v) {
    const int k = ((v >> 2) << 4) | (khalf << 3) | ((v & 3) << 1);
    r.u[v] = p[k >> 1];
  }
  return r.v;
}

// C/D layout: lanes 0-15: N=lane, M=vgpr; lanes 16-31: N=lane-16, M=vgpr+8
__device__ __forceinline__ void store_tile_f32(float* C, int ldc, int m0, int n0,
                                               int wm, int wn, int lane,
                                               const v8f acc[2][4]) {
#pragma unroll
  for (int mi = 0; mi < 2; ++mi)
#pragma unroll
    for (int ni = 0; ni < 4; ++ni) {
      const int row0 = m0 + wm * 32 + mi * 16 + ((lane >> 4) << 3);
      const int col = n0 + wn * 64 + ni * 16 + (lane & 15);
#pragma unroll
      for (int r = 0; r < 8; ++r)
        C[(size_t)(row0 + r) * ldc + col] = acc[mi][ni][r];
    }
}

// ---------------- Stage 1: H = Kmat(S,S) @ X(S,D), fp32 in -> fp16 out ---------
__global__ __launch_bounds__(256) void gemm_pre_kernel(
    const float* __restrict__ Kmat, const float* __restrict__ x1,
    const float* __restrict__ x2, _Float16* __restrict__ h1,
    _Float16* __restrict__ h2) {
  __shared__ alignas(16) _Float16 As[2][BM * LDA];
  __shared__ alignas(16) _Float16 Bst[2][BN * LDB];
  const int n0 = blockIdx.x * BN;
  const int m0 = blockIdx.y * BM;
  const int b = blockIdx.z >> 1;
  const int which = blockIdx.z & 1;
  const float* __restrict__ X = (which ? x2 : x1) + (size_t)b * S_DIM * D_DIM;
  _Float16* __restrict__ H = (which ? h2 : h1) + (size_t)b * S_DIM * D_DIM;

  const int tid = threadIdx.x;
  const int lane = tid & 31;
  const int wave = tid >> 5;
  const int wm = wave >> 1, wn = wave & 1;

  // copy-thread coordinates
  const int ra = tid >> 1, ca = (tid & 1) << 4;        // A: 128 rows x 2 float4-pairs
  const int rb = tid >> 5, cb = (tid & 31) << 2;       // B: 32 rows x 32 float4

  float4 stA[4], stB[4];
  auto fetch = [&](int k0) {
#pragma unroll
    for (int it = 0; it < 4; ++it)   // A: Kmat[m0+r][k0+c] 128x32 f32
      stA[it] = *reinterpret_cast<const float4*>(
          Kmat + (size_t)(m0 + ra) * S_DIM + k0 + ca + ((it & 1) << 2) +
          ((it >> 1) << 3));
#pragma unroll
    for (int it = 0; it < 4; ++it)   // B: X[k0+r][n0+c] 32x128 f32
      stB[it] = *reinterpret_cast<const float4*>(
          X + (size_t)(k0 + rb + (it << 3)) * D_DIM + n0 + cb);
    if (k0 + BK < S_DIM)
      __builtin_prefetch(Kmat + (size_t)(m0 + ra) * S_DIM + k0 + BK + ca, 0, 1);
  };
  auto commit = [&](_Float16* Ad, _Float16* Bd) {
#pragma unroll
    for (int it = 0; it < 4; ++it) {
      H4 cv;
      cv.h[0] = (_Float16)stA[it].x; cv.h[1] = (_Float16)stA[it].y;
      cv.h[2] = (_Float16)stA[it].z; cv.h[3] = (_Float16)stA[it].w;
      *reinterpret_cast<uint2*>(Ad + ra * LDA + ca + ((it & 1) << 2) +
                                ((it >> 1) << 3)) = cv.u;
    }
#pragma unroll
    for (int it = 0; it < 4; ++it) {
      const int r = rb + (it << 3);
      Bd[(cb + 0) * LDB + r] = (_Float16)stB[it].x;
      Bd[(cb + 1) * LDB + r] = (_Float16)stB[it].y;
      Bd[(cb + 2) * LDB + r] = (_Float16)stB[it].z;
      Bd[(cb + 3) * LDB + r] = (_Float16)stB[it].w;
    }
  };

  v8f acc[2][4] = {};
  fetch(0);
  commit(As[0], Bst[0]);
  int cur = 0;
  for (int k0 = 0; k0 < S_DIM; k0 += BK) {
    __syncthreads();
    const _Float16* cA = As[cur];
    const _Float16* cB = Bst[cur];
    v16h a[2], bfr[4];
    a[0] = load_frag_a(cA, wm * 32, lane);
    a[1] = load_frag_a(cA, wm * 32 + 16, lane);
#pragma unroll
    for (int ni = 0; ni < 4; ++ni) bfr[ni] = load_frag_b(cB, wn * 64 + ni * 16, lane);
    const bool more = (k0 + BK) < S_DIM;
    if (more) fetch(k0 + BK);
#pragma unroll
    for (int mi = 0; mi < 2; ++mi)
#pragma unroll
      for (int ni = 0; ni < 4; ++ni)
        acc[mi][ni] = wmma_f16(a[mi], bfr[ni], acc[mi][ni]);
    if (more) commit(As[cur ^ 1], Bst[cur ^ 1]);
    cur ^= 1;
  }
#pragma unroll
  for (int mi = 0; mi < 2; ++mi)
#pragma unroll
    for (int ni = 0; ni < 4; ++ni) {
      const int row0 = m0 + wm * 32 + mi * 16 + ((lane >> 4) << 3);
      const int col = n0 + wn * 64 + ni * 16 + (lane & 15);
#pragma unroll
      for (int r = 0; r < 8; ++r)
        H[(size_t)(row0 + r) * D_DIM + col] = (_Float16)acc[mi][ni][r];
    }
}

// ---------------- Stage 2: scores = h1 @ h2^T, causal tiles only --------------
__global__ __launch_bounds__(256) void gemm_scores_kernel(
    const _Float16* __restrict__ h1, const _Float16* __restrict__ h2,
    float* __restrict__ scores) {
  const int t0 = blockIdx.x * BN;
  const int s0 = blockIdx.y * BM;
  if (t0 > s0 + BM - 1) return;   // fully masked tile: softmax never reads t>s
  __shared__ alignas(16) _Float16 As[2][BM * LDA];
  __shared__ alignas(16) _Float16 Bst[2][BN * LDB];
  const _Float16* __restrict__ A = h1 + (size_t)blockIdx.z * S_DIM * D_DIM;
  const _Float16* __restrict__ Bm = h2 + (size_t)blockIdx.z * S_DIM * D_DIM;
  float* __restrict__ C = scores + (size_t)blockIdx.z * S_DIM * S_DIM;

  const int tid = threadIdx.x;
  const int lane = tid & 31;
  const int wave = tid >> 5;
  const int wm = wave >> 1, wn = wave & 1;

  const int r8 = tid >> 1, c8 = (tid & 1) << 4;  // 128 rows x 2 (8-half chunks)

  uint4 stA[2], stB[2];
  auto fetch = [&](int k0) {
#pragma unroll
    for (int it = 0; it < 2; ++it) {  // A: h1[s0+r][k0+c] 128x32 f16
      stA[it] = *reinterpret_cast<const uint4*>(
          A + (size_t)(s0 + r8) * D_DIM + k0 + c8 + (it << 3));
      stB[it] = *reinterpret_cast<const uint4*>(
          Bm + (size_t)(t0 + r8) * D_DIM + k0 + c8 + (it << 3));
    }
    if (k0 + BK < D_DIM)
      __builtin_prefetch(A + (size_t)(s0 + r8) * D_DIM + k0 + BK + c8, 0, 1);
  };
  auto commit = [&](_Float16* Ad, _Float16* Bd) {
#pragma unroll
    for (int it = 0; it < 2; ++it) {
      // 8-byte LDS stores (rows are 8B-aligned, not 16B)
      *reinterpret_cast<uint2*>(Ad + r8 * LDA + c8 + (it << 3)) =
          make_uint2(stA[it].x, stA[it].y);
      *reinterpret_cast<uint2*>(Ad + r8 * LDA + c8 + (it << 3) + 4) =
          make_uint2(stA[it].z, stA[it].w);
      *reinterpret_cast<uint2*>(Bd + r8 * LDB + c8 + (it << 3)) =
          make_uint2(stB[it].x, stB[it].y);
      *reinterpret_cast<uint2*>(Bd + r8 * LDB + c8 + (it << 3) + 4) =
          make_uint2(stB[it].z, stB[it].w);
    }
  };

  v8f acc[2][4] = {};
  fetch(0);
  commit(As[0], Bst[0]);
  int cur = 0;
  for (int k0 = 0; k0 < D_DIM; k0 += BK) {
    __syncthreads();
    const _Float16* cA = As[cur];
    const _Float16* cB = Bst[cur];
    v16h a[2], bfr[4];
    a[0] = load_frag_a(cA, wm * 32, lane);
    a[1] = load_frag_a(cA, wm * 32 + 16, lane);
#pragma unroll
    for (int ni = 0; ni < 4; ++ni) bfr[ni] = load_frag_b(cB, wn * 64 + ni * 16, lane);
    const bool more = (k0 + BK) < D_DIM;
    if (more) fetch(k0 + BK);
#pragma unroll
    for (int mi = 0; mi < 2; ++mi)
#pragma unroll
      for (int ni = 0; ni < 4; ++ni)
        acc[mi][ni] = wmma_f16(a[mi], bfr[ni], acc[mi][ni]);
    if (more) commit(As[cur ^ 1], Bst[cur ^ 1]);
    cur ^= 1;
  }
  store_tile_f32(C, S_DIM, s0, t0, wm, wn, lane, acc);
}

// ---------------- Stage 3: causal row softmax (with 1/sqrt(S) scale) ----------
__global__ __launch_bounds__(256) void softmax_kernel(
    const float* __restrict__ scores, _Float16* __restrict__ P) {
  __shared__ float red[256];
  const int row = blockIdx.x;  // b*S + s
  const int s = row & (S_DIM - 1);
  const float* __restrict__ src = scores + (size_t)row * S_DIM;
  _Float16* __restrict__ dst = P + (size_t)row * S_DIM;
  const int tid = threadIdx.x;
  constexpr float inv_dk = 0.022097086912079608f;  // 1/sqrt(2048)

  float v[8];
  float mx = -3.0e38f;
#pragma unroll
  for (int i = 0; i < 8; ++i) {
    const int t = tid + (i << 8);
    v[i] = (t <= s) ? src[t] * inv_dk : -3.0e38f;
    mx = fmaxf(mx, v[i]);
  }
  red[tid] = mx;
  __syncthreads();
#pragma unroll
  for (int o = 128; o > 0; o >>= 1) {
    if (tid < o) red[tid] = fmaxf(red[tid], red[tid + o]);
    __syncthreads();
  }
  mx = red[0];
  __syncthreads();

  float sum = 0.f;
#pragma unroll
  for (int i = 0; i < 8; ++i) {
    const int t = tid + (i << 8);
    const float e = (t <= s) ? __expf(v[i] - mx) : 0.f;
    v[i] = e;
    sum += e;
  }
  red[tid] = sum;
  __syncthreads();
#pragma unroll
  for (int o = 128; o > 0; o >>= 1) {
    if (tid < o) red[tid] += red[tid + o];
    __syncthreads();
  }
  const float inv = 1.0f / red[0];
#pragma unroll
  for (int i = 0; i < 8; ++i)
    dst[tid + (i << 8)] = (_Float16)(v[i] * inv);  // t>s written as exact 0
}

// ---------------- Stage 4: out = P @ h1, causal K-limit -----------------------
__global__ __launch_bounds__(256) void gemm_out_kernel(
    const _Float16* __restrict__ P, const _Float16* __restrict__ h1,
    float* __restrict__ out) {
  __shared__ alignas(16) _Float16 As[2][BM * LDA];
  __shared__ alignas(16) _Float16 Bst[2][BN * LDB];
  const int n0 = blockIdx.x * BN;  // d
  const int s0 = blockIdx.y * BM;  // s
  const int b = blockIdx.z;
  const _Float16* __restrict__ A = P + (size_t)b * S_DIM * S_DIM;
  const _Float16* __restrict__ Bm = h1 + (size_t)b * S_DIM * D_DIM;
  float* __restrict__ C = out + (size_t)b * S_DIM * D_DIM;

  const int tid = threadIdx.x;
  const int lane = tid & 31;
  const int wave = tid >> 5;
  const int wm = wave >> 1, wn = wave & 1;

  const int kend = s0 + BM;  // P[s][t]==0 for t>s

  const int ra = tid >> 1, ca = (tid & 1) << 4;   // A: 128 rows x 2 chunks of 8
  const int rb = tid >> 5, cb = (tid & 31) << 2;  // B: 32 rows x 32 uint2

  uint4 stA[2];
  uint2 stB[4];
  auto fetch = [&](int k0) {
#pragma unroll
    for (int it = 0; it < 2; ++it)  // A: P[s0+r][k0+c] 128x32 f16, stride S
      stA[it] = *reinterpret_cast<const uint4*>(
          A + (size_t)(s0 + ra) * S_DIM + k0 + ca + (it << 3));
#pragma unroll
    for (int it = 0; it < 4; ++it)  // B: h1[k0+r][n0+c] 32x128 f16
      stB[it] = *reinterpret_cast<const uint2*>(
          Bm + (size_t)(k0 + rb + (it << 3)) * D_DIM + n0 + cb);
    if (k0 + BK < kend)
      __builtin_prefetch(A + (size_t)(s0 + ra) * S_DIM + k0 + BK + ca, 0, 1);
  };
  auto commit = [&](_Float16* Ad, _Float16* Bd) {
#pragma unroll
    for (int it = 0; it < 2; ++it) {
      *reinterpret_cast<uint2*>(Ad + ra * LDA + ca + (it << 3)) =
          make_uint2(stA[it].x, stA[it].y);
      *reinterpret_cast<uint2*>(Ad + ra * LDA + ca + (it << 3) + 4) =
          make_uint2(stA[it].z, stA[it].w);
    }
#pragma unroll
    for (int it = 0; it < 4; ++it) {  // transpose into Bd[n][k]
      const int r = rb + (it << 3);
      H4 cv;
      cv.u = stB[it];
      Bd[(cb + 0) * LDB + r] = cv.h[0];
      Bd[(cb + 1) * LDB + r] = cv.h[1];
      Bd[(cb + 2) * LDB + r] = cv.h[2];
      Bd[(cb + 3) * LDB + r] = cv.h[3];
    }
  };

  v8f acc[2][4] = {};
  fetch(0);
  commit(As[0], Bst[0]);
  int cur = 0;
  for (int k0 = 0; k0 < kend; k0 += BK) {
    __syncthreads();
    const _Float16* cA = As[cur];
    const _Float16* cB = Bst[cur];
    v16h a[2], bfr[4];
    a[0] = load_frag_a(cA, wm * 32, lane);
    a[1] = load_frag_a(cA, wm * 32 + 16, lane);
#pragma unroll
    for (int ni = 0; ni < 4; ++ni) bfr[ni] = load_frag_b(cB, wn * 64 + ni * 16, lane);
    const bool more = (k0 + BK) < kend;
    if (more) fetch(k0 + BK);
#pragma unroll
    for (int mi = 0; mi < 2; ++mi)
#pragma unroll
      for (int ni = 0; ni < 4; ++ni)
        acc[mi][ni] = wmma_f16(a[mi], bfr[ni], acc[mi][ni]);
    if (more) commit(As[cur ^ 1], Bst[cur ^ 1]);
    cur ^= 1;
  }
  store_tile_f32(C, D_DIM, s0, n0, wm, wn, lane, acc);
}

extern "C" void kernel_launch(void* const* d_in, const int* in_sizes, int n_in,
                              void* d_out, int out_size, void* d_ws, size_t ws_size,
                              hipStream_t stream) {
  const float* x1 = (const float*)d_in[0];
  const float* x2 = (const float*)d_in[1];
  const float* Kmat = (const float*)d_in[2];
  float* out = (float*)d_out;

  // Workspace: h1,h2 fp16 (16 MiB each) | scores fp32 (128 MiB) | P fp16 (64 MiB)
  const size_t Hbytes = (size_t)B_DIM * S_DIM * D_DIM * sizeof(_Float16);
  const size_t SCbytes = (size_t)B_DIM * S_DIM * S_DIM * sizeof(float);
  char* w = (char*)d_ws;
  _Float16* h1 = (_Float16*)w;
  _Float16* h2 = (_Float16*)(w + Hbytes);
  float* scores = (float*)(w + 2 * Hbytes);
  _Float16* P = (_Float16*)(w + 2 * Hbytes + SCbytes);

  gemm_pre_kernel<<<dim3(D_DIM / BN, S_DIM / BM, 2 * B_DIM), 256, 0, stream>>>(
      Kmat, x1, x2, h1, h2);
  gemm_scores_kernel<<<dim3(S_DIM / BN, S_DIM / BM, B_DIM), 256, 0, stream>>>(
      h1, h2, scores);
  softmax_kernel<<<B_DIM * S_DIM, 256, 0, stream>>>(scores, P);
  gemm_out_kernel<<<dim3(D_DIM / BN, S_DIM / BM, B_DIM), 256, 0, stream>>>(
      P, h1, out);
}
